// KoLeoLoss_5643587027141
// MI455X (gfx1250) — matrix operand order repaired and checked by reading, not compiled, not run on hardware
//
#include <hip/hip_runtime.h>

typedef _Float16 f16;
typedef _Float16 v4h  __attribute__((ext_vector_type(4)));
typedef _Float16 v8h  __attribute__((ext_vector_type(8)));
typedef _Float16 v16h __attribute__((ext_vector_type(16)));
typedef float    v8f  __attribute__((ext_vector_type(8)));
typedef int      v4i  __attribute__((ext_vector_type(4)));

#define DIM 1024
#define KOLEO_EPS 1e-8f
#define MB 64                 // rows (M) per block: A strip = 64x1024 f16 in LDS
#define APAD 8                // LDS row pad (halves): 2064B stride -> 4-bank skew
#define ASTRIDE (DIM + APAD)

// ---------------------------------------------------------------------------
// Kernel 1: per-row L2 normalize -> fp32 copy (distance pass) + f16 copy
// (WMMA Gram matrix). One 256-thread block per row.
// ---------------------------------------------------------------------------
__global__ void __launch_bounds__(256)
koleo_normalize(const float* __restrict__ x,
                float* __restrict__ xn,
                f16* __restrict__ xnh) {
  const int row = blockIdx.x;
  const int tid = threadIdx.x;
  const float4 v = ((const float4*)(x + (size_t)row * DIM))[tid];
  float s = v.x * v.x + v.y * v.y + v.z * v.z + v.w * v.w;
  for (int off = 16; off; off >>= 1) s += __shfl_down(s, off, 32);
  __shared__ float red[8];
  __shared__ float sscale;
  if ((tid & 31) == 0) red[tid >> 5] = s;
  __syncthreads();
  if (tid == 0) {
    float t = 0.0f;
    for (int i = 0; i < 8; ++i) t += red[i];
    sscale = 1.0f / fmaxf(sqrtf(t), KOLEO_EPS);
  }
  __syncthreads();
  const float sc = sscale;
  float4 o;
  o.x = v.x * sc; o.y = v.y * sc; o.z = v.z * sc; o.w = v.w * sc;
  ((float4*)(xn + (size_t)row * DIM))[tid] = o;
  v4h h;
  h[0] = (f16)o.x; h[1] = (f16)o.y; h[2] = (f16)o.z; h[3] = (f16)o.w;
  ((v4h*)(xnh + (size_t)row * DIM))[tid] = h;
}

// Monotone float -> u32 map (order-preserving incl. negatives).
__device__ __forceinline__ unsigned int fmono(float f) {
  unsigned int u = __float_as_uint(f);
  return (u & 0x80000000u) ? ~u : (u | 0x80000000u);
}

// ---------------------------------------------------------------------------
// Kernel 2: fused Gram-matrix + per-row argmax (diagonal masked).
// Block = 64 rows; A strip staged in LDS via async global->LDS (padded
// stride -> conflict-free ds_load_b128). 8 waves x 16 cols = 128 columns per
// iteration; each wave reuses its B fragment across 4 M-subtiles
// (4 x v_wmma_f32_16x16x32_f16 per B load). Per-lane running (key,idx)
// best is updated branchlessly inside the column loop (v_cndmask only, no
// exec juggling) and co-executes with the matrix pipe; the 4-step butterfly
// + ds_max_u64 merge runs once at the end.
// ---------------------------------------------------------------------------
__global__ void __launch_bounds__(256)
koleo_nn_search(const f16* __restrict__ xnh, int* __restrict__ nnidx, int B) {
  __shared__ __align__(16) f16 Atile[MB * ASTRIDE];        // ~129 KB of 320 KB
  __shared__ unsigned long long best[MB];

  const int tid  = threadIdx.x;
  const int lane = tid & 31;
  const int wave = tid >> 5;
  const int rowBase = blockIdx.x * MB;

  // Stage the 64xDIM A strip: 8192 x 16B chunks, async direct-to-LDS.
  {
    const int nch = MB * (DIM / 8);
    for (int c = tid; c < nch; c += 256) {
      const int r   = c >> 7;            // DIM/8 = 128 chunks per row
      const int col = (c & 127) * 8;
      const f16* g = xnh + (size_t)(rowBase + r) * DIM + col;
      f16* l = &Atile[r * ASTRIDE + col];
#if __has_builtin(__builtin_amdgcn_global_load_async_to_lds_b128)
      __builtin_amdgcn_global_load_async_to_lds_b128(
          (__attribute__((address_space(1))) v4i*)g,
          (__attribute__((address_space(3))) v4i*)l, 0, 0);
#else
      *(uint4*)l = *(const uint4*)g;
#endif
    }
#if __has_builtin(__builtin_amdgcn_global_load_async_to_lds_b128)
#if __has_builtin(__builtin_amdgcn_s_wait_asynccnt)
    __builtin_amdgcn_s_wait_asynccnt(0);
#else
    asm volatile("s_wait_asynccnt 0" ::: "memory");
#endif
#endif
  }
  for (int i = tid; i < MB; i += 256) best[i] = 0ull;
  __syncthreads();

  const int hi  = lane >> 4;       // half-wave select per ISA fragment layout
  const int l15 = lane & 15;

  union V16 { v16h v; struct { v8h lo, hi; } p; };

  // Per-lane running argmax state: monotone key + column index per row-slot.
  unsigned int bKey[4][8];
  unsigned int bIdx[4][8];
#pragma unroll
  for (int m = 0; m < 4; ++m)
#pragma unroll
    for (int r = 0; r < 8; ++r) { bKey[m][r] = 0u; bIdx[m][r] = 0u; }

  for (int iter = 0; iter < B / 128; ++iter) {
    const int nCol = iter * 128 + wave * 16 + l15;   // global column index
    const f16* bp = xnh + (size_t)nCol * DIM;
    v8f acc[4] = {{}, {}, {}, {}};
    for (int k0 = 0; k0 < DIM; k0 += 32) {
      // B 32x16 fragment: lane's column = row nCol of xnh (contiguous 32B).
      const v16h b = *(const v16h*)(bp + k0 + hi * 16);
#pragma unroll
      for (int m = 0; m < 4; ++m) {
        // A 16x32 fragment for M-subtile m: lanes 0-15 K={k0..+15},
        // lanes 16-31 K={k0+16..+31} (matches B). Padded stride -> no bank
        // conflicts across the 16 rows.
        const f16* ap = &Atile[(m * 16 + l15) * ASTRIDE + k0 + hi * 16];
        V16 a;
        a.p.lo = *(const v8h*)ap;
        a.p.hi = *(const v8h*)(ap + 8);
        acc[m] = __builtin_amdgcn_wmma_f32_16x16x32_f16(
            false, a.v, false, b, (short)0, acc[m], false, false);
      }
    }
    // Branchless running-best update. C layout: lane holds column N = l15,
    // rows M = m*16 + r + hi*8. Diagonal -> key 0 (never wins).
#pragma unroll
    for (int m = 0; m < 4; ++m) {
#pragma unroll
      for (int r = 0; r < 8; ++r) {
        const int grow = rowBase + m * 16 + r + hi * 8;
        unsigned int key = fmono(acc[m][r]);
        key = (grow == nCol) ? 0u : key;
        const bool gt = key > bKey[m][r];
        bKey[m][r] = gt ? key  : bKey[m][r];
        bIdx[m][r] = gt ? (unsigned int)nCol : bIdx[m][r];
      }
    }
  }

  // One butterfly argmax over the 16 columns per half-wave, then merge the
  // 8 waves' results via ds_max_u64 on packed (key<<32 | idx).
#pragma unroll
  for (int m = 0; m < 4; ++m) {
#pragma unroll
    for (int r = 0; r < 8; ++r) {
      unsigned int key = bKey[m][r];
      unsigned int idx = bIdx[m][r];
#pragma unroll
      for (int msk = 1; msk <= 8; msk <<= 1) {
        const unsigned int ok = (unsigned int)__shfl_xor((int)key, msk, 32);
        const unsigned int oi = (unsigned int)__shfl_xor((int)idx, msk, 32);
        const bool gt = ok > key;
        key = gt ? ok : key;
        idx = gt ? oi : idx;
      }
      if (l15 == 0) {
        unsigned long long pk = (((unsigned long long)key) << 32) | idx;
        atomicMax(&best[m * 16 + r + hi * 8], pk);      // ds_max_u64
      }
    }
  }
  __syncthreads();
  if (tid < MB) nnidx[rowBase + tid] = (int)(best[tid] & 0xFFFFFFFFu);
}

// ---------------------------------------------------------------------------
// Kernel 3: per-row ||xn[i] - xn[nn[i]] + eps||_2 -> log(dist + eps).
// ---------------------------------------------------------------------------
__global__ void __launch_bounds__(256)
koleo_dist(const float* __restrict__ xn, const int* __restrict__ nnidx,
           float* __restrict__ logs) {
  const int row = blockIdx.x;
  const int tid = threadIdx.x;
  const int nn  = nnidx[row];
  const float4 va = ((const float4*)(xn + (size_t)row * DIM))[tid];
  const float4 vb = ((const float4*)(xn + (size_t)nn  * DIM))[tid];
  const float dx = va.x - vb.x + KOLEO_EPS;
  const float dy = va.y - vb.y + KOLEO_EPS;
  const float dz = va.z - vb.z + KOLEO_EPS;
  const float dw = va.w - vb.w + KOLEO_EPS;
  float s = dx * dx + dy * dy + dz * dz + dw * dw;
  for (int off = 16; off; off >>= 1) s += __shfl_down(s, off, 32);
  __shared__ float red[8];
  if ((tid & 31) == 0) red[tid >> 5] = s;
  __syncthreads();
  if (tid == 0) {
    float t = 0.0f;
    for (int i = 0; i < 8; ++i) t += red[i];
    logs[row] = logf(sqrtf(t) + KOLEO_EPS);
  }
}

// ---------------------------------------------------------------------------
// Kernel 4: deterministic single-block reduction -> loss = -mean(logs).
// ---------------------------------------------------------------------------
__global__ void __launch_bounds__(256)
koleo_finalize(const float* __restrict__ logs, float* __restrict__ out, int B) {
  const int tid = threadIdx.x;
  float s = 0.0f;
  for (int i = tid; i < B; i += 256) s += logs[i];
  for (int off = 16; off; off >>= 1) s += __shfl_down(s, off, 32);
  __shared__ float red[8];
  if ((tid & 31) == 0) red[tid >> 5] = s;
  __syncthreads();
  if (tid == 0) {
    float t = 0.0f;
    for (int i = 0; i < 8; ++i) t += red[i];
    out[0] = -t / (float)B;
  }
}

// ---------------------------------------------------------------------------
extern "C" void kernel_launch(void* const* d_in, const int* in_sizes, int n_in,
                              void* d_out, int out_size, void* d_ws,
                              size_t ws_size, hipStream_t stream) {
  (void)n_in; (void)out_size; (void)ws_size;
  const float* x = (const float*)d_in[0];
  const int B = in_sizes[0] / DIM;                 // 8192

  char* ws = (char*)d_ws;
  float* xn    = (float*)ws;                                        // B*DIM f32
  f16*   xnh   = (f16*)(ws + (size_t)B * DIM * 4);                  // B*DIM f16
  int*   nnidx = (int*)(ws + (size_t)B * DIM * 6);                  // B ints
  float* logs  = (float*)(ws + (size_t)B * DIM * 6 + (size_t)B * 4);// B floats
  float* out   = (float*)d_out;

  koleo_normalize<<<B, 256, 0, stream>>>(x, xn, xnh);
  koleo_nn_search<<<B / MB, 256, 0, stream>>>(xnh, nnidx, B);
  koleo_dist<<<B, 256, 0, stream>>>(xn, nnidx, logs);
  koleo_finalize<<<1, 256, 0, stream>>>(logs, out, B);
}
